// kronecker_Net_3_6296422056379
// MI455X (gfx1250) — matrix operand order repaired and checked by reading, not compiled, not run on hardware
//
#include <hip/hip_runtime.h>
#include <hip/hip_bf16.h>

#define NN 100000
#define NE 1600000
#define NG 1024
#define BN_EPS 1e-5f

typedef __attribute__((ext_vector_type(16))) __bf16 v16bf;
typedef __attribute__((ext_vector_type(8)))  __bf16 v8bf;
typedef __attribute__((ext_vector_type(8)))  float  v8f;

__device__ __forceinline__ unsigned short bf16bits(float f) {
  unsigned u = __float_as_uint(f);
  u += 0x7FFFu + ((u >> 16) & 1u);           // round-to-nearest-even
  return (unsigned short)(u >> 16);
}

// ---------------- elementwise helpers ----------------
__global__ void k_copy(float* __restrict__ dst, const float* __restrict__ src, int n) {
  for (int i = blockIdx.x * blockDim.x + threadIdx.x; i < n; i += gridDim.x * blockDim.x)
    dst[i] = src[i];
}

__global__ void k_zero(float* __restrict__ dst, int n) {
  for (int i = blockIdx.x * blockDim.x + threadIdx.x; i < n; i += gridDim.x * blockDim.x)
    dst[i] = 0.0f;
}

// acc[dst[e]] += h[src[e]]   (one block per edge; L2-resident float atomics)
__global__ void k_edge_add(const float* __restrict__ h, const int* __restrict__ esrc,
                           const int* __restrict__ edst, int dim, float* __restrict__ acc) {
  int e = blockIdx.x;
  int k = threadIdx.x;
  if (k >= dim) return;
  int s = esrc[e], d = edst[e];
  atomicAdd(&acc[d * dim + k], h[s * dim + k]);
}

// fp32 [M,K] -> bf16 [M,Kpad] row-major, zero padded; optional fused BN+ReLU.
template <bool BN>
__global__ void k_conv_bf16(const float* __restrict__ x, int K, int Kpad,
                            const float* __restrict__ scale, const float* __restrict__ shift,
                            unsigned short* __restrict__ y, int M) {
  int total = M * Kpad;
  for (int i = blockIdx.x * blockDim.x + threadIdx.x; i < total; i += gridDim.x * blockDim.x) {
    int m = i / Kpad, k = i - m * Kpad;
    float v = 0.0f;
    if (k < K) {
      v = x[m * K + k];
      if (BN) v = fmaxf(scale[k] * v + shift[k], 0.0f);
    }
    y[i] = bf16bits(v);
  }
}

// Pack fp32 weights [K,N] into WMMA B-fragment order: [nt][ks][lane][e] bf16,
// where n = nt*16 + (lane&15), k = ks*32 + (lane>>4)*8 + (e&7) + (e>>3)*16.
__global__ void k_pack_w(const float* __restrict__ W, int K, int N, int Ksteps, int Ntiles,
                         unsigned short* __restrict__ out) {
  int total = Ntiles * Ksteps * 32 * 16;
  for (int i = blockIdx.x * blockDim.x + threadIdx.x; i < total; i += gridDim.x * blockDim.x) {
    int e  = i & 15;
    int l  = (i >> 4) & 31;
    int r  = i >> 9;
    int ks = r % Ksteps;
    int nt = r / Ksteps;
    int n = (nt << 4) + (l & 15);
    int k = (ks << 5) + ((l >> 4) << 3) + (e & 7) + ((e >> 3) << 4);
    float v = (k < K && n < N) ? W[k * N + n] : 0.0f;
    out[i] = bf16bits(v);
  }
}

// ---------------- WMMA GEMM: out = [relu]( A_bf16 @ Bpacked + bias ) ----------------
// A: bf16 [M, Kpad] row-major (pre-transformed).  Bp: packed fragments.
// Inner loop: 4x global_load_b128 + 1x v_wmma. One 16x16 tile per wave.
__global__ void k_gemm_wmma(const unsigned short* __restrict__ A, int Kpad,
                            const unsigned short* __restrict__ Bp,
                            const float* __restrict__ bias,
                            float* __restrict__ out, int ldo,
                            int M, int N, int relu_out) {
  const int lane = threadIdx.x & 31;
  const int wave = threadIdx.x >> 5;
  const int lr   = lane & 15;
  const int hi   = lane >> 4;
  const int Mtiles = M >> 4;                 // M is a multiple of 16
  const int Ntiles = (N + 15) >> 4;
  int t = blockIdx.x * (blockDim.x >> 5) + wave;   // uniform per wave
  if (t >= Mtiles * Ntiles) return;
  const int mt = t / Ntiles, nt = t % Ntiles;
  const int Ksteps = Kpad >> 5;

  const unsigned short* ap = A + (long)((mt << 4) + lr) * Kpad + hi * 8;
  const unsigned short* bp = Bp + ((long)nt * Ksteps * 32 + lane) * 16;

  v8f c = {};
#pragma unroll 4
  for (int ks = 0; ks < Ksteps; ++ks) {
    v8bf a0 = *(const v8bf*)(ap);            // K block [k0+hi*8 .. +8)
    v8bf a1 = *(const v8bf*)(ap + 16);       // K block [k0+16+hi*8 .. +8)
    v16bf a = __builtin_shufflevector(a0, a1, 0, 1, 2, 3, 4, 5, 6, 7,
                                             8, 9, 10, 11, 12, 13, 14, 15);
    v16bf b = *(const v16bf*)(bp);
    c = __builtin_amdgcn_wmma_f32_16x16x32_bf16(false, a, false, b, (short)0, c, false, false);
    ap += 32;
    bp += 32 * 16;
  }

  const int bcol = (nt << 4) + lr;
  if (bcol < N) {
    float bv = bias[bcol];
#pragma unroll
    for (int r = 0; r < 8; ++r) {
      int row = (mt << 4) + (hi ? 8 + r : r);
      float v = c[r] + bv;
      if (relu_out) v = fmaxf(v, 0.0f);
      out[row * ldo + bcol] = v;
    }
  }
}

// ---------------- training-mode BN: column sums / sums of squares ----------------
__global__ void k_col_stats(const float* __restrict__ h, int M, int N,
                            float* __restrict__ sums, float* __restrict__ sq) {
  __shared__ float ss[128];
  __shared__ float s2[128];
  for (int c = threadIdx.x; c < N; c += blockDim.x) { ss[c] = 0.0f; s2[c] = 0.0f; }
  __syncthreads();
  int total = M * N;
  for (int i = blockIdx.x * blockDim.x + threadIdx.x; i < total; i += gridDim.x * blockDim.x) {
    float v = h[i];
    int c = i % N;
    atomicAdd(&ss[c], v);
    atomicAdd(&s2[c], v * v);
  }
  __syncthreads();
  for (int c = threadIdx.x; c < N; c += blockDim.x) {
    atomicAdd(&sums[c], ss[c]);
    atomicAdd(&sq[c],  s2[c]);
  }
}

__global__ void k_make_bn(const float* __restrict__ sums, const float* __restrict__ sq, float M,
                          const float* __restrict__ gamma, const float* __restrict__ beta,
                          float* __restrict__ scale, float* __restrict__ shift, int N) {
  int c = blockIdx.x * blockDim.x + threadIdx.x;
  if (c >= N) return;
  float mean = sums[c] / M;
  float var  = sq[c] / M - mean * mean;
  float sc   = gamma[c] * rsqrtf(var + BN_EPS);
  scale[c] = sc;
  shift[c] = beta[c] - mean * sc;
}

__global__ void k_bn_relu(const float* __restrict__ x, const float* __restrict__ scale,
                          const float* __restrict__ shift, float* __restrict__ y, int n, int N) {
  for (int i = blockIdx.x * blockDim.x + threadIdx.x; i < n; i += gridDim.x * blockDim.x) {
    int c = i % N;
    y[i] = fmaxf(scale[c] * x[i] + shift[c], 0.0f);
  }
}

// ---------------- graph mean pooling ----------------
__global__ void k_graph_cnt(const int* __restrict__ n2g, int n, float* __restrict__ cnt) {
  for (int i = blockIdx.x * blockDim.x + threadIdx.x; i < n; i += gridDim.x * blockDim.x)
    atomicAdd(&cnt[n2g[i]], 1.0f);
}

__global__ void k_graph_sum(const float* __restrict__ h, const int* __restrict__ n2g,
                            int n, int dim, float* __restrict__ gsum) {
  int total = n * dim;
  for (int i = blockIdx.x * blockDim.x + threadIdx.x; i < total; i += gridDim.x * blockDim.x) {
    int node = i / dim, c = i % dim;
    atomicAdd(&gsum[n2g[node] * dim + c], h[i]);
  }
}

__global__ void k_hg(const float* __restrict__ gsum, const float* __restrict__ cnt,
                     float* __restrict__ hg, int total, int dim) {
  for (int i = blockIdx.x * blockDim.x + threadIdx.x; i < total; i += gridDim.x * blockDim.x)
    hg[i] = gsum[i] / fmaxf(cnt[i / dim], 1.0f);
}

// ---------------- head: kron fusion + fc1 ----------------
__global__ void k_kron_fc1(const float* __restrict__ hg, const float* __restrict__ self,
                           const float* __restrict__ W, const float* __restrict__ bias,
                           float* __restrict__ out) {
  __shared__ float sh[20];
  __shared__ float ss[16];
  int g = blockIdx.x, t = threadIdx.x;        // blockDim = 32
  if (t < 20) sh[t] = hg[g * 20 + t];
  if (t < 16) ss[t] = self[g * 16 + t];
  __syncthreads();
  float acc = bias[t];
  for (int i = 0; i < 20; ++i) {
    float hv = sh[i];
#pragma unroll
    for (int j = 0; j < 16; ++j)
      acc += hv * ss[j] * W[(i * 16 + j) * 32 + t];
  }
  out[g * 32 + t] = acc;
}

__global__ void k_fc(const float* __restrict__ x, const float* __restrict__ W,
                     const float* __restrict__ bias, float* __restrict__ y,
                     int M, int K, int N) {
  int total = M * N;
  for (int i = blockIdx.x * blockDim.x + threadIdx.x; i < total; i += gridDim.x * blockDim.x) {
    int m = i / N, n = i % N;
    float acc = bias[n];
    for (int k = 0; k < K; ++k) acc += x[m * K + k] * W[k * N + n];
    y[i] = acc;
  }
}

// ---------------- host orchestration ----------------
extern "C" void kernel_launch(void* const* d_in, const int* in_sizes, int n_in,
                              void* d_out, int out_size, void* d_ws, size_t ws_size,
                              hipStream_t stream) {
  const float* feat      = (const float*)d_in[0];
  const float* self_feat = (const float*)d_in[1];
  const int*   esrc      = (const int*)d_in[2];
  const int*   edst      = (const int*)d_in[3];
  const int*   n2g       = (const int*)d_in[4];
  const float* g1_W1 = (const float*)d_in[5],  *g1_b1 = (const float*)d_in[6];
  const float* g1_g  = (const float*)d_in[7],  *g1_be = (const float*)d_in[8];
  const float* g1_W2 = (const float*)d_in[9],  *g1_b2 = (const float*)d_in[10];
  const float* g2_W1 = (const float*)d_in[11], *g2_b1 = (const float*)d_in[12];
  const float* g2_g  = (const float*)d_in[13], *g2_be = (const float*)d_in[14];
  const float* g2_W2 = (const float*)d_in[15], *g2_b2 = (const float*)d_in[16];
  const float* fc1_W = (const float*)d_in[17], *fc1_b = (const float*)d_in[18];
  const float* bn1_g = (const float*)d_in[19], *bn1_b = (const float*)d_in[20];
  const float* fc2_W = (const float*)d_in[21], *fc2_b = (const float*)d_in[22];
  const float* bn2_g = (const float*)d_in[23], *bn2_b = (const float*)d_in[24];
  const float* fc3_W = (const float*)d_in[25], *fc3_b = (const float*)d_in[26];
  float* out = (float*)d_out;

  float* ws   = (float*)d_ws;
  float* bufA = ws + 0;             // 12,800,000 f: x0+agg, later x1+agg
  float* bufB = ws + 12800000;      // 10,000,000 f: h1_pre, later h2_pre
  float* bufC = ws + 22800000;      // 10,000,000 f: x1, later h_final
  unsigned short* Abf = (unsigned short*)(ws + 32800000);  // 12,800,000 bf16 (6.4M f)
  float* S    = ws + 39200000;      // small region
  float* sums  = S + 0;       float* sq    = S + 128;
  float* scale = S + 256;     float* shift = S + 384;
  float* gsum  = S + 512;     /* 20480 */
  float* gcnt  = S + 20992;   /* 1024  */
  float* hg    = S + 22016;   /* 20480 */
  float* pre1  = S + 42496;   /* 32768 */
  float* act1  = S + 75264;   /* 32768 */
  float* pre2  = S + 108032;  /* 8192  */
  float* act2  = S + 116224;  /* 8192  */
  unsigned short* pw = (unsigned short*)(S + 124416);
  unsigned short* pw1 = pw + 0;       // 7*4*512 = 14336
  unsigned short* pw2 = pw + 14336;   // 7*4*512 = 14336
  unsigned short* pw3 = pw + 28672;   // 2*4*512 = 4096
  unsigned short* pw4 = pw + 32768;   // 2*1*512 = 1024

  auto gemm = [&](const unsigned short* A, int Kpad, const unsigned short* Bp,
                  const float* bias, float* o, int ldo, int M, int N, int relu) {
    int tiles = (M / 16) * ((N + 15) / 16);
    int blocks = (tiles + 3) / 4;
    k_gemm_wmma<<<blocks, 128, 0, stream>>>(A, Kpad, Bp, bias, o, ldo, M, N, relu);
  };

  // ---- pack all GEMM weights into WMMA B-fragment layout (tiny, one pass) ----
  k_pack_w<<<16, 256, 0, stream>>>(g1_W1, 128, 100, 4, 7, pw1);
  k_pack_w<<<16, 256, 0, stream>>>(g1_W2, 100, 100, 4, 7, pw2);
  k_pack_w<<<16, 256, 0, stream>>>(g2_W1, 100, 20, 4, 2, pw3);
  k_pack_w<<<16, 256, 0, stream>>>(g2_W2, 20, 20, 1, 2, pw4);

  // ---- GIN layer 1 ----
  k_copy<<<4096, 256, 0, stream>>>(bufA, feat, NN * 128);
  k_edge_add<<<NE, 128, 0, stream>>>(feat, esrc, edst, 128, bufA);
  k_conv_bf16<false><<<4096, 256, 0, stream>>>(bufA, 128, 128, nullptr, nullptr, Abf, NN);
  gemm(Abf, 128, pw1, g1_b1, bufB, 100, NN, 100, 0);
  k_zero<<<1, 256, 0, stream>>>(sums, 256);
  k_col_stats<<<512, 256, 0, stream>>>(bufB, NN, 100, sums, sq);
  k_make_bn<<<1, 128, 0, stream>>>(sums, sq, (float)NN, g1_g, g1_be, scale, shift, 100);
  k_conv_bf16<true><<<4096, 256, 0, stream>>>(bufB, 100, 128, scale, shift, Abf, NN);
  gemm(Abf, 128, pw2, g1_b2, bufC, 100, NN, 100, 1);

  // ---- GIN layer 2 ----
  k_copy<<<4096, 256, 0, stream>>>(bufA, bufC, NN * 100);
  k_edge_add<<<NE, 128, 0, stream>>>(bufC, esrc, edst, 100, bufA);
  k_conv_bf16<false><<<4096, 256, 0, stream>>>(bufA, 100, 128, nullptr, nullptr, Abf, NN);
  gemm(Abf, 128, pw3, g2_b1, bufB, 20, NN, 20, 0);
  k_zero<<<1, 256, 0, stream>>>(sums, 256);
  k_col_stats<<<512, 256, 0, stream>>>(bufB, NN, 20, sums, sq);
  k_make_bn<<<1, 128, 0, stream>>>(sums, sq, (float)NN, g2_g, g2_be, scale, shift, 20);
  k_conv_bf16<true><<<1024, 256, 0, stream>>>(bufB, 20, 32, scale, shift, Abf, NN);
  gemm(Abf, 32, pw4, g2_b2, bufC, 20, NN, 20, 1);

  // ---- per-graph mean pooling ----
  k_zero<<<64, 256, 0, stream>>>(gsum, 21504);   // gsum + gcnt (contiguous)
  k_graph_cnt<<<256, 256, 0, stream>>>(n2g, NN, gcnt);
  k_graph_sum<<<1024, 256, 0, stream>>>(bufC, n2g, NN, 20, gsum);
  k_hg<<<80, 256, 0, stream>>>(gsum, gcnt, hg, NG * 20, 20);

  // ---- head: kron + fc1 + BN + relu ----
  k_kron_fc1<<<NG, 32, 0, stream>>>(hg, self_feat, fc1_W, fc1_b, pre1);
  k_zero<<<1, 256, 0, stream>>>(sums, 256);
  k_col_stats<<<64, 256, 0, stream>>>(pre1, NG, 32, sums, sq);
  k_make_bn<<<1, 128, 0, stream>>>(sums, sq, (float)NG, bn1_g, bn1_b, scale, shift, 32);
  k_bn_relu<<<128, 256, 0, stream>>>(pre1, scale, shift, act1, NG * 32, 32);

  // ---- fc2 + BN + relu ----
  k_fc<<<32, 256, 0, stream>>>(act1, fc2_W, fc2_b, pre2, NG, 32, 8);
  k_zero<<<1, 256, 0, stream>>>(sums, 256);
  k_col_stats<<<32, 256, 0, stream>>>(pre2, NG, 8, sums, sq);
  k_make_bn<<<1, 128, 0, stream>>>(sums, sq, (float)NG, bn2_g, bn2_b, scale, shift, 8);
  k_bn_relu<<<32, 256, 0, stream>>>(pre2, scale, shift, act2, NG * 8, 8);

  // ---- fc3 -> d_out [1024] ----
  k_fc<<<4, 256, 0, stream>>>(act2, fc3_W, fc3_b, out, NG, 8, 1);
}
